// AttentionFlow_Layer_335007449489
// MI455X (gfx1250) — matrix operand order repaired and checked by reading, not compiled, not run on hardware
//
#include <hip/hip_runtime.h>

// ---------------------------------------------------------------------------
// Problem constants (match reference)
// ---------------------------------------------------------------------------
#define B_   8
#define LC_  2048
#define LQ_  128
#define D_   256     // 2*MD
#define D8_  1024    // 8*MD (feature / hidden width)
#define NEGV (-10000000000.0f)

typedef __attribute__((ext_vector_type(16))) _Float16 v16h;
typedef __attribute__((ext_vector_type(8)))  _Float16 v8h;
typedef __attribute__((ext_vector_type(2)))  _Float16 v2h;
typedef __attribute__((ext_vector_type(8)))  float    v8f;
typedef int b128i __attribute__((vector_size(16)));   // 16-byte chunk type

#ifndef __has_builtin
#define __has_builtin(x) 0
#endif

// ---------------------------------------------------------------------------
// CDNA5 async global->LDS copy (ASYNCcnt path), with sync fallback
// Builtin signature (from hipcc diagnostic): param0 = AS(1) b128i*,
// param1 = AS(3) b128i*, then imm offset, imm cpol.
// ---------------------------------------------------------------------------
#if __has_builtin(__builtin_amdgcn_global_load_async_to_lds_b128)
#define HAVE_ASYNC_COPY 1
#else
#define HAVE_ASYNC_COPY 0
#endif

#if HAVE_ASYNC_COPY
#if __has_builtin(__builtin_amdgcn_s_wait_asynccnt)
#define WAIT_ASYNC(n) __builtin_amdgcn_s_wait_asynccnt((unsigned short)(n))
#else
#define WAIT_ASYNC(n) asm volatile("s_wait_asynccnt %0" ::"i"(n) : "memory")
#endif
#else
#define WAIT_ASYNC(n) ((void)0)
#endif

static __device__ __forceinline__ void copy16B_to_lds(_Float16* dst_lds,
                                                      const _Float16* src_g) {
#if HAVE_ASYNC_COPY
  __builtin_amdgcn_global_load_async_to_lds_b128(
      (__attribute__((address_space(1))) b128i*)src_g,
      (__attribute__((address_space(3))) b128i*)dst_lds, 0, 0);
#else
  *(v8h*)dst_lds = *(const v8h*)src_g;
#endif
}

// ---------------------------------------------------------------------------
// WMMA helpers (CDNA5 16x16x32 f16 -> f32). C/D layout per ISA 7.12.2; A/B use contiguous k = 16*(lane/16) + s.
// ---------------------------------------------------------------------------
static __device__ __forceinline__ v8f wmma_f16(v16h a, v16h b, v8f c) {
  // 8 args: (neg_a, A, neg_b, B, c_mod, C, reuse_a, reuse_b)
  return __builtin_amdgcn_wmma_f32_16x16x32_f16(false, a, false, b, (short)0, c,
                                                false, false);
}

// A fragment: 16x32 (MxK), memory row-major [row][k], row stride = ld halfs.
static __device__ __forceinline__ v16h load_a_frag(const _Float16* base, int ld,
                                                   int row0, int k0, int lane) {
  const int grp = lane >> 4;
  const _Float16* p = base + (row0 + (lane & 15)) * ld + k0 + grp * 16;
  v8h lo = *(const v8h*)(p);
  v8h hi = *(const v8h*)(p + 8);
  v16h r;
#pragma unroll
  for (int i = 0; i < 8; ++i) { r[i] = lo[i]; r[8 + i] = hi[i]; }
  return r;
}

// B fragment: 32x16 (KxN) where memory holds B transposed: mem[n][k].
static __device__ __forceinline__ v16h load_bt_frag(const _Float16* base, int ld,
                                                    int n0, int k0, int lane) {
  const int grp = lane >> 4;
  const _Float16* p = base + (n0 + (lane & 15)) * ld + k0 + grp * 16;
  v8h lo = *(const v8h*)(p);
  v8h hi = *(const v8h*)(p + 8);
  v16h r;
#pragma unroll
  for (int i = 0; i < 8; ++i) { r[i] = lo[i]; r[8 + i] = hi[i]; }
  return r;
}

// B fragment from natural [k][n] memory (gather per lane).
static __device__ __forceinline__ v16h load_b_frag(const _Float16* base, int ld,
                                                   int k0, int n0, int lane) {
  const int grp = lane >> 4;
  const int n = n0 + (lane & 15);
  v16h r;
#pragma unroll
  for (int kk = 0; kk < 16; ++kk) r[kk] = base[(k0 + grp * 16 + kk) * ld + n];
  return r;
}

// ---------------------------------------------------------------------------
// Kernel 1: convert + TRANSPOSE w_inner / w_out fp32 -> f16  (W^T, [n][k])
// Transposed layout makes GEMM B-tile staging a straight contiguous copy
// (async-to-LDS friendly) and keeps B fragments as vector ds_load_b128.
// ---------------------------------------------------------------------------
__global__ __launch_bounds__(256) void wcvt_kernel(const float* __restrict__ wi,
                                                   const float* __restrict__ wo,
                                                   _Float16* __restrict__ wiT,
                                                   _Float16* __restrict__ woT) {
  size_t idx = (size_t)blockIdx.x * 256 + threadIdx.x;  // 0 .. 1M-1
  const int k = (int)(idx >> 10);
  const int n = (int)(idx & 1023);
  wiT[(size_t)n * D8_ + k] = (_Float16)wi[idx];
  woT[(size_t)n * D8_ + k] = (_Float16)wo[idx];
}

// ---------------------------------------------------------------------------
// Kernel 2: fused sim + masked softmax + c2q for one (b, 128-context-row) tile
// ---------------------------------------------------------------------------
__global__ __launch_bounds__(256) void sim_c2q_kernel(
    const float* __restrict__ ctx, const unsigned char* __restrict__ cmask,
    const float* __restrict__ qry, const unsigned char* __restrict__ qmask,
    const float* __restrict__ wsim, const float* __restrict__ bsimp,
    float* __restrict__ mrow, float* __restrict__ c2q) {
  __shared__ _Float16 Qs[LQ_ * D_];         // 64 KB: query tile f16 [j][d]
  __shared__ _Float16 As[128 * D_];         // 64 KB: scaled ctx f16, reused as P
  __shared__ float scS[128];
  __shared__ float sqS[128];
  __shared__ unsigned char cmS[128];
  __shared__ unsigned char qmS[128];

  const int b   = blockIdx.y;
  const int i0  = blockIdx.x * 128;
  const int tid = threadIdx.x;
  const int lane = tid & 31;
  const int w    = tid >> 5;        // wave id, 8 waves (wave32)
  const int grp  = lane >> 4;

  const float* wc = wsim;
  const float* wq = wsim + D_;
  const float* wx = wsim + 2 * D_;
  const float bsim = bsimp[0];

  // ---- stage Q (f16) + sq + query mask --------------------------------
  {
    const int row = tid >> 1;
    const int cb  = (tid & 1) * 128;
    const float* src = qry + ((size_t)b * LQ_ + row) * D_ + cb;
    float pd = 0.f;
#pragma unroll 8
    for (int c = 0; c < 128; c += 2) {
      float2 v = *(const float2*)(src + c);
      pd += v.x * wq[cb + c] + v.y * wq[cb + c + 1];
      v2h h; h[0] = (_Float16)v.x; h[1] = (_Float16)v.y;
      *(v2h*)&Qs[row * D_ + cb + c] = h;
    }
    pd += __shfl_xor(pd, 1);
    if ((tid & 1) == 0) sqS[row] = pd;
    if (tid < LQ_) qmS[tid] = qmask[b * LQ_ + tid];
  }
  // ---- stage A = ctx*wx (f16) + sc + context mask ---------------------
  {
    const int row = tid >> 1;
    const int cb  = (tid & 1) * 128;
    const float* src = ctx + ((size_t)b * LC_ + i0 + row) * D_ + cb;
    float pd = 0.f;
#pragma unroll 8
    for (int c = 0; c < 128; c += 2) {
      float2 v = *(const float2*)(src + c);
      pd += v.x * wc[cb + c] + v.y * wc[cb + c + 1];
      v2h h;
      h[0] = (_Float16)(v.x * wx[cb + c]);
      h[1] = (_Float16)(v.y * wx[cb + c + 1]);
      *(v2h*)&As[row * D_ + cb + c] = h;
    }
    pd += __shfl_xor(pd, 1);
    if ((tid & 1) == 0) scS[row] = pd;
    if (tid < 128) cmS[tid] = cmask[b * LC_ + i0 + tid];
  }
  __syncthreads();

  // ---- S = A @ Q^T : wave w owns rows [16w, 16w+16), all 128 cols -----
  v8f acc[8];
  {
    v8f z = {};
#pragma unroll
    for (int nt = 0; nt < 8; ++nt) acc[nt] = z;
  }
  for (int k0 = 0; k0 < D_; k0 += 32) {
    v16h a = load_a_frag(As, D_, 16 * w, k0, lane);
#pragma unroll
    for (int nt = 0; nt < 8; ++nt) {
      // S = A * Q^T: Q row-major [j][d] is exactly B^T layout
      v16h bf = load_bt_frag(Qs, D_, nt * 16, k0, lane);
      acc[nt] = wmma_f16(a, bf, acc[nt]);
    }
  }

  // ---- epilogue: add sc/sq/bias, mask (C layout: lane<16 M=r, else M=8+r)
#pragma unroll
  for (int nt = 0; nt < 8; ++nt) {
#pragma unroll
    for (int r = 0; r < 8; ++r) {
      const int row = 16 * w + r + 8 * grp;
      const int col = nt * 16 + (lane & 15);
      float s = acc[nt][r] + scS[row] + sqS[col] + bsim;
      if (cmS[row] | qmS[col]) s = NEGV;
      acc[nt][r] = s;
    }
  }

  // ---- row max over j (16-lane group reduce), store for q2c -----------
  float rmax[8];
#pragma unroll
  for (int r = 0; r < 8; ++r) {
    float m = -3.4e38f;
#pragma unroll
    for (int nt = 0; nt < 8; ++nt) m = fmaxf(m, acc[nt][r]);
    m = fmaxf(m, __shfl_xor(m, 1));
    m = fmaxf(m, __shfl_xor(m, 2));
    m = fmaxf(m, __shfl_xor(m, 4));
    m = fmaxf(m, __shfl_xor(m, 8));
    rmax[r] = m;
  }
  if ((lane & 15) == 0) {
#pragma unroll
    for (int r = 0; r < 8; ++r)
      mrow[(size_t)b * LC_ + i0 + 16 * w + r + 8 * grp] = rmax[r];
  }

  // ---- exp + row sum --------------------------------------------------
  float rinv[8];
#pragma unroll
  for (int r = 0; r < 8; ++r) {
    float s = 0.f;
#pragma unroll
    for (int nt = 0; nt < 8; ++nt) {
      float e = __expf(acc[nt][r] - rmax[r]);
      acc[nt][r] = e;
      s += e;
    }
    s += __shfl_xor(s, 1);
    s += __shfl_xor(s, 2);
    s += __shfl_xor(s, 4);
    s += __shfl_xor(s, 8);
    rinv[r] = 1.0f / s;
  }

  // ---- write normalized P (f16) into As region (reuse) ----------------
  __syncthreads();                       // everyone done reading As
  _Float16* Ps = As;                     // [128][128]
#pragma unroll
  for (int nt = 0; nt < 8; ++nt) {
#pragma unroll
    for (int r = 0; r < 8; ++r) {
      const int row = 16 * w + r + 8 * grp;
      const int col = nt * 16 + (lane & 15);
      Ps[row * LQ_ + col] = (_Float16)(acc[nt][r] * rinv[r]);
    }
  }
  __syncthreads();

  // ---- c2q = P @ Q : N = 256 -> 16 column tiles, K = 128 --------------
  for (int nt = 0; nt < 16; ++nt) {
    v8f aq = {};
    for (int k0 = 0; k0 < LQ_; k0 += 32) {
      v16h a  = load_a_frag(Ps, LQ_, 16 * w, k0, lane);
      v16h bf = load_b_frag(Qs, D_, k0, nt * 16, lane);  // B[k=j][n=d]=Q[j][d]
      aq = wmma_f16(a, bf, aq);
    }
#pragma unroll
    for (int r = 0; r < 8; ++r) {
      const int row = i0 + 16 * w + r + 8 * grp;
      const int col = nt * 16 + (lane & 15);
      c2q[((size_t)b * LC_ + row) * D_ + col] = aq[r];
    }
  }
}

// ---------------------------------------------------------------------------
// Kernel 3: q2c[b,d] = sum_i softmax_i(mrow[b,i]) * ctx[b,i,d]
// ---------------------------------------------------------------------------
__global__ __launch_bounds__(256) void q2c_kernel(const float* __restrict__ ctx,
                                                  const float* __restrict__ mrow,
                                                  float* __restrict__ q2c) {
  __shared__ float red[256];
  __shared__ float wts[LC_];   // 8 KB
  const int b = blockIdx.x, tid = threadIdx.x;

  float lmax = -3.4e38f;
  for (int i = tid; i < LC_; i += 256) lmax = fmaxf(lmax, mrow[b * LC_ + i]);
  red[tid] = lmax;
  __syncthreads();
  for (int s = 128; s > 0; s >>= 1) {
    if (tid < s) red[tid] = fmaxf(red[tid], red[tid + s]);
    __syncthreads();
  }
  const float M = red[0];
  __syncthreads();

  float lsum = 0.f;
  for (int i = tid; i < LC_; i += 256) {
    float e = __expf(mrow[b * LC_ + i] - M);
    wts[i] = e;
    lsum += e;
  }
  red[tid] = lsum;
  __syncthreads();
  for (int s = 128; s > 0; s >>= 1) {
    if (tid < s) red[tid] += red[tid + s];
    __syncthreads();
  }
  const float Zi = 1.0f / red[0];
  __syncthreads();

  const int d = tid;   // D_ == blockDim == 256
  float a = 0.f;
  for (int i = 0; i < LC_; ++i)
    a += wts[i] * ctx[((size_t)b * LC_ + i) * D_ + d];
  q2c[b * D_ + d] = a * Zi;
}

// ---------------------------------------------------------------------------
// Kernel 4: assemble feat = [ctx | c2q | ctx*c2q | ctx*q2c]  (f16, width 1024)
// ---------------------------------------------------------------------------
__global__ __launch_bounds__(256) void feat_kernel(const float* __restrict__ ctx,
                                                   const float* __restrict__ c2q,
                                                   const float* __restrict__ q2c,
                                                   _Float16* __restrict__ feat) {
  const size_t g = blockIdx.x;          // 0 .. B*LC-1
  const int d = threadIdx.x;            // 0 .. 255
  const int b = (int)(g >> 11);
  const float cv = ctx[g * D_ + d];
  const float c2 = c2q[g * D_ + d];
  const float qv = q2c[b * D_ + d];
  _Float16* f = feat + g * D8_;
  f[d]            = (_Float16)cv;
  f[D_ + d]       = (_Float16)c2;
  f[2 * D_ + d]   = (_Float16)(cv * c2);
  f[3 * D_ + d]   = (_Float16)(cv * qv);
}

// ---------------------------------------------------------------------------
// Kernel 5/6: C = act(A_f16[M x 1024] @ W[1024 x 1024] + bias), W given as W^T.
// Double-buffered async global->LDS staging (ASYNCcnt), 128x128 tile / block,
// 8 waves, wave strip = 16 rows.
//   outH != nullptr -> ReLU + f16 store (hidden)
//   outH == nullptr -> fp32 store to d_out, zero where cmask[row]
// ---------------------------------------------------------------------------
__global__ __launch_bounds__(256) void mlp_gemm_kernel(
    const _Float16* __restrict__ A, const _Float16* __restrict__ WT,
    const float* __restrict__ bias, const unsigned char* __restrict__ cmask,
    _Float16* __restrict__ outH, float* __restrict__ outF, int relu) {
  __shared__ _Float16 At[2][128 * 32];   // 2 x 8 KB  [row][k]
  __shared__ _Float16 Bt[2][128 * 32];   // 2 x 8 KB  [n][k]  (W^T tile)
  __shared__ float biasS[128];

  const int n0 = blockIdx.x * 128;
  const int m0 = blockIdx.y * 128;
  const int tid = threadIdx.x;
  const int lane = tid & 31;
  const int w = tid >> 5;
  const int grp = lane >> 4;

  if (tid < 128) biasS[tid] = bias[n0 + tid];

  // 16B chunk geometry: 8 KB tile = 512 chunks; this thread owns chunks
  // tid and tid+256.  chunk c -> row c>>2, halfs offset (c&3)*8.
  const int ra = tid >> 2,          sa = (tid & 3) * 8;
  const int rb = (tid + 256) >> 2,  sb = ((tid + 256) & 3) * 8;

  v8f acc[8];
  {
    v8f z = {};
#pragma unroll
    for (int nt = 0; nt < 8; ++nt) acc[nt] = z;
  }

  // issue one tile stage (4 async-copy instructions per wave)
  auto stage = [&](int buf, int k0) {
    copy16B_to_lds(&At[buf][ra * 32 + sa], A  + (size_t)(m0 + ra) * D8_ + k0 + sa);
    copy16B_to_lds(&At[buf][rb * 32 + sb], A  + (size_t)(m0 + rb) * D8_ + k0 + sb);
    copy16B_to_lds(&Bt[buf][ra * 32 + sa], WT + (size_t)(n0 + ra) * D8_ + k0 + sa);
    copy16B_to_lds(&Bt[buf][rb * 32 + sb], WT + (size_t)(n0 + rb) * D8_ + k0 + sb);
  };

  stage(0, 0);
  const int NT = D8_ / 32;  // 32 k-tiles
  for (int kt = 0; kt < NT; ++kt) {
    __syncthreads();   // all waves done reading the buffer we are about to fill
    if (kt + 1 < NT) {
      stage((kt + 1) & 1, (kt + 1) * 32);  // prefetch next tile
      WAIT_ASYNC(4);                       // current tile done, prefetch in flight
    } else {
      WAIT_ASYNC(0);
    }
    __syncthreads();   // current tile visible to all waves

    const _Float16* at = At[kt & 1];
    const _Float16* bt = Bt[kt & 1];
    v16h a = load_a_frag(at, 32, 16 * w, 0, lane);
#pragma unroll
    for (int nt = 0; nt < 8; ++nt) {
      v16h bf = load_bt_frag(bt, 32, nt * 16, 0, lane);
      acc[nt] = wmma_f16(a, bf, acc[nt]);
    }
  }

  // epilogue
#pragma unroll
  for (int nt = 0; nt < 8; ++nt) {
    const int col = n0 + nt * 16 + (lane & 15);
#pragma unroll
    for (int r = 0; r < 8; ++r) {
      const int row = m0 + 16 * w + r + 8 * grp;
      float v = acc[nt][r] + biasS[nt * 16 + (lane & 15)];
      if (relu) v = fmaxf(v, 0.f);
      if (outH != nullptr) {
        outH[(size_t)row * D8_ + col] = (_Float16)v;
      } else {
        if (cmask[row]) v = 0.f;
        outF[(size_t)row * D8_ + col] = v;
      }
    }
  }
}

// ---------------------------------------------------------------------------
// Host launcher. Workspace layout (~88.2 MB needed):
//   mrow   : B*LC fp32                (65536 B)
//   q2c    : B*D fp32                 (8192 B)
//   c2q    : B*LC*D fp32              (16.78 MB)
//   wInT   : 1024*1024 f16 (W^T)      (2.10 MB)
//   wOutT  : 1024*1024 f16 (W^T)      (2.10 MB)
//   featH  : B*LC*1024 f16            (33.55 MB)
//   hidH   : B*LC*1024 f16            (33.55 MB)
// ---------------------------------------------------------------------------
extern "C" void kernel_launch(void* const* d_in, const int* in_sizes, int n_in,
                              void* d_out, int out_size, void* d_ws,
                              size_t ws_size, hipStream_t stream) {
  const float* ctx           = (const float*)d_in[0];
  const unsigned char* cmask = (const unsigned char*)d_in[1];  // bool array
  const float* qry           = (const float*)d_in[2];
  const unsigned char* qmask = (const unsigned char*)d_in[3];  // bool array
  const float* wsim          = (const float*)d_in[4];
  const float* bsim          = (const float*)d_in[5];
  const float* w_inner       = (const float*)d_in[6];
  const float* b_inner       = (const float*)d_in[7];
  const float* w_out         = (const float*)d_in[8];
  const float* b_out         = (const float*)d_in[9];
  float* out                 = (float*)d_out;

  char* ws = (char*)d_ws;
  float* mrow = (float*)ws;                 ws += (size_t)B_ * LC_ * 4;
  float* q2c  = (float*)ws;                 ws += (size_t)B_ * D_ * 4;
  float* c2q  = (float*)ws;                 ws += (size_t)B_ * LC_ * D_ * 4;
  _Float16* wInT  = (_Float16*)ws;          ws += (size_t)D8_ * D8_ * 2;
  _Float16* wOutT = (_Float16*)ws;          ws += (size_t)D8_ * D8_ * 2;
  _Float16* featH = (_Float16*)ws;          ws += (size_t)B_ * LC_ * D8_ * 2;
  _Float16* hidH  = (_Float16*)ws;          ws += (size_t)B_ * LC_ * D8_ * 2;

  // 1) weights -> f16, transposed
  wcvt_kernel<<<(D8_ * D8_) / 256, 256, 0, stream>>>(w_inner, w_out, wInT, wOutT);

  // 2) fused sim / softmax / c2q (+ rowmax)
  sim_c2q_kernel<<<dim3(LC_ / 128, B_), 256, 0, stream>>>(
      ctx, cmask, qry, qmask, wsim, bsim, mrow, c2q);

  // 3) q2c
  q2c_kernel<<<B_, 256, 0, stream>>>(ctx, mrow, q2c);

  // 4) feature assembly (f16)
  feat_kernel<<<B_ * LC_, 256, 0, stream>>>(ctx, c2q, q2c, featH);

  // 5) hidden = relu(feat @ w_inner + b_inner)   (f16 out)
  mlp_gemm_kernel<<<dim3(D8_ / 128, (B_ * LC_) / 128), 256, 0, stream>>>(
      featH, wInT, b_inner, nullptr, hidH, nullptr, 1);

  // 6) out = (hidden @ w_out + b_out), zeroed where context_mask (fp32 out)
  mlp_gemm_kernel<<<dim3(D8_ / 128, (B_ * LC_) / 128), 256, 0, stream>>>(
      hidH, wOutT, b_out, cmask, nullptr, out, 0);
}
